// MultiHeadDotProductAttentionWithPrefix_73856257622021
// MI455X (gfx1250) — matrix operand order repaired and checked by reading, not compiled, not run on hardware
//
#include <hip/hip_runtime.h>
#include <hip/hip_bf16.h>

// MHA with prefix KV for gfx1250 (MI455X), wave32 + v_wmma_f32_16x16x32_f16.
// B=2, LQ=LKV=2048, E=1024, H=16, D=64, P=64, KT=2112.

typedef _Float16 h4   __attribute__((ext_vector_type(4)));
typedef _Float16 h8   __attribute__((ext_vector_type(8)));
typedef _Float16 v16h __attribute__((ext_vector_type(16)));
typedef float    v8f  __attribute__((ext_vector_type(8)));
typedef float    f4   __attribute__((ext_vector_type(4)));

#define NEG_BIAS (-1e10f)

// Load a 16-element f16 fragment as two contiguous 16-byte chunks.
__device__ __forceinline__ v16h ld_frag(const _Float16* lo_p, const _Float16* hi_p) {
    h8 lo = *(const h8*)lo_p;
    h8 hi = *(const h8*)hi_p;
    v16h r;
#pragma unroll
    for (int i = 0; i < 8; ++i) { r[i] = lo[i]; r[i + 8] = hi[i]; }
    return r;
}

__device__ __forceinline__ v8f wmma16(v16h a, v16h b, v8f c) {
    // (neg_a, A, neg_b, B, c_mod, C, reuse_a, reuse_b)
    return __builtin_amdgcn_wmma_f32_16x16x32_f16(false, a, false, b, (short)0, c,
                                                  false, false);
}

// ---------------------------------------------------------------------------
// GEMM  C[M=4096, N=1024] = A[M,1024] * W[1024,1024], templated on MODE:
// MODE 0: A=inputs_q  f32, W=Wq, out -> Qh  [B,H,LQ,D] f16, scaled 1/sqrt(64)
// MODE 1: A=inputs_kv f32, W=Wk, out -> Kh  [B,H,KT,D] f16 (offset P rows)
// MODE 2: A=inputs_kv f32, W=Wv, out -> Vth [B,H,D,KT] f16 (transposed)
// MODE 3: A=Ctx       f16, W=Wo, out -> out f32 [B,LQ,E]
// Block: 256 threads (8 waves), 128x64 C tile, K-chunks of 32 through LDS.
// ---------------------------------------------------------------------------
template <int MODE>
__global__ __launch_bounds__(256) void gemm_qkvo(
    const float* __restrict__ A32, const _Float16* __restrict__ A16,
    const float* __restrict__ W, _Float16* __restrict__ dstH,
    float* __restrict__ dstF) {
    __shared__ _Float16 As[128][40];  // 128 M-rows x 32 K (pad 8 -> rows 16B aligned)
    __shared__ _Float16 WsT[64][40];  // 64 N-rows x 32 K (transposed W tile)

    const int tid  = threadIdx.x;
    const int lane = tid & 31;
    const int wave = tid >> 5;
    const int m0   = blockIdx.x * 128;
    const int n0   = blockIdx.y * 64;
    const int colr = lane & 15;
    const int halfu = lane >> 4;

    v8f acc[4];
#pragma unroll
    for (int t = 0; t < 4; ++t) acc[t] = (v8f){0, 0, 0, 0, 0, 0, 0, 0};

    for (int kc = 0; kc < 32; ++kc) {
        const int kb = kc * 32;
        // ---- stage A tile (128 x 32) ----
        if (MODE == 3) {
            // f16 source: pure b128 copies (512 h8 groups, 2 per thread).
#pragma unroll
            for (int i = 0; i < 2; ++i) {
                const int idx = tid + i * 256;          // 0..511
                const int r = idx >> 2, c8 = idx & 3;   // 4 h8-groups per row
                h8 v = *(const h8*)&A16[(size_t)(m0 + r) * 1024 + kb + c8 * 8];
                *(h8*)&As[r][c8 * 8] = v;
            }
        } else {
            // f32 source: float4 load -> 4x cvt -> ds_store_b64 (1024 groups, 4/thread).
#pragma unroll
            for (int i = 0; i < 4; ++i) {
                const int idx = tid + i * 256;          // 0..1023
                const int r = idx >> 3, c4 = idx & 7;   // 8 float4-groups per row
                f4 v = *(const f4*)&A32[(size_t)(m0 + r) * 1024 + kb + c4 * 4];
                h4 hv;
#pragma unroll
                for (int j = 0; j < 4; ++j) hv[j] = (_Float16)v[j];
                *(h4*)&As[r][c4 * 4] = hv;
            }
        }
        // ---- stage W tile transposed: WsT[n][k] = W[kb+k][n0+n] ----
        // 512 h4 groups (c = n-col 0..63, k4 = k-quad 0..7); coalesced b32 reads.
#pragma unroll
        for (int i = 0; i < 2; ++i) {
            const int idx = tid + i * 256;              // 0..511
            const int c = idx & 63, k4 = idx >> 6;
            h4 hv;
#pragma unroll
            for (int j = 0; j < 4; ++j)
                hv[j] = (_Float16)W[(size_t)(kb + k4 * 4 + j) * 1024 + n0 + c];
            *(h4*)&WsT[c][k4 * 4] = hv;
        }
        __syncthreads();

        // A fragment (16x32): lane<16 holds K 0..15; lane>=16 holds 16..31
        const _Float16* ap = &As[wave * 16 + colr][0];
        v16h afrag = ld_frag(ap + halfu * 16, ap + halfu * 16 + 8);
#pragma unroll
        for (int t = 0; t < 4; ++t) {
            // B fragment (32x16): lane holds col n%16, K = (lane>=16?16:0)+0..15
            const _Float16* bp = &WsT[t * 16 + colr][0] + halfu * 16;
            v16h bfrag = ld_frag(bp, bp + 8);
            acc[t] = wmma16(afrag, bfrag, acc[t]);
        }
        __syncthreads();
    }

    // Scatter C (C-layout: VGPR r, lane n -> M = r + 8*(n>=16), N = n%16).
#pragma unroll
    for (int t = 0; t < 4; ++t) {
        const int ncol = n0 + t * 16 + colr;
#pragma unroll
        for (int r = 0; r < 8; ++r) {
            const int m = m0 + wave * 16 + r + (halfu << 3);
            const float val = acc[t][r];
            if (MODE == 3) {
                dstF[(size_t)m * 1024 + ncol] = val;
            } else {
                const int b = m >> 11, l = m & 2047;
                const int h = ncol >> 6, d = ncol & 63;
                if (MODE == 0)
                    dstH[(((size_t)(b * 16 + h)) * 2048 + l) * 64 + d] =
                        (_Float16)(val * 0.125f);  // 1/sqrt(D)
                else if (MODE == 1)
                    dstH[(((size_t)(b * 16 + h)) * 2112 + 64 + l) * 64 + d] =
                        (_Float16)val;
                else
                    dstH[(((size_t)(b * 16 + h)) * 64 + d) * 2112 + 64 + l] =
                        (_Float16)val;
            }
        }
    }
}

// ---------------------------------------------------------------------------
// Copy prefix K/V (f32 [B,P,H,D]) into f16 Kh / transposed Vth.
// ---------------------------------------------------------------------------
__global__ __launch_bounds__(256) void prefix_copy(
    const float* __restrict__ kpre, const float* __restrict__ vpre,
    _Float16* __restrict__ Kh, _Float16* __restrict__ Vth) {
    const int idx = blockIdx.x * 256 + threadIdx.x;  // B*P*H*D = 131072
    if (idx >= 2 * 64 * 16 * 64) return;
    const int d = idx & 63;
    const int h = (idx >> 6) & 15;
    const int p = (idx >> 10) & 63;
    const int b = idx >> 16;
    const size_t src = (((size_t)(b * 64 + p)) * 16 + h) * 64 + d;
    Kh[(((size_t)(b * 16 + h)) * 2112 + p) * 64 + d] = (_Float16)kpre[src];
    Vth[(((size_t)(b * 16 + h)) * 64 + d) * 2112 + p] = (_Float16)vpre[src];
}

// ---------------------------------------------------------------------------
// Flash attention. Block = 128 threads (4 waves); wave owns 16 q-rows x D=64.
// Per 32-col KV step: 4 WMMA for S, online softmax, 4 WMMA for P*V.
// Causal bias computed analytically (prefix cols 0..63 always attendable).
// ---------------------------------------------------------------------------
__global__ __launch_bounds__(128) void flash_attn(
    const _Float16* __restrict__ Qh, const _Float16* __restrict__ Kh,
    const _Float16* __restrict__ Vth, _Float16* __restrict__ Ctx) {
    __shared__ _Float16 Pt[4][16][40];  // per-wave 16x32 prob tile (pad 8)

    const int lane = threadIdx.x & 31;
    const int wave = threadIdx.x >> 5;
    const int h = blockIdx.y, b = blockIdx.z;
    const int bh = b * 16 + h;
    const int qrow0 = blockIdx.x * 64 + wave * 16;
    const int colr = lane & 15;
    const int halfu = lane >> 4;

    // Q fragments for the two 32-wide D chunks (A-layout, loaded once).
    const _Float16* qp = Qh + (((size_t)bh) * 2048 + qrow0 + colr) * 64;
    const v16h aq0 = ld_frag(qp + halfu * 16, qp + halfu * 16 + 8);
    const v16h aq1 = ld_frag(qp + 32 + halfu * 16, qp + 40 + halfu * 16);

    float mstat[8], lstat[8];
    v8f o[4];
#pragma unroll
    for (int r = 0; r < 8; ++r) { mstat[r] = -1e30f; lstat[r] = 0.0f; }
#pragma unroll
    for (int t = 0; t < 4; ++t) o[t] = (v8f){0, 0, 0, 0, 0, 0, 0, 0};

    const _Float16* Kb = Kh + (size_t)bh * 2112 * 64;
    const _Float16* Vb = Vth + (size_t)bh * 64 * 2112;

    int kv_end = (qrow0 + 64 + 16 + 31) & ~31;  // prefix + last q row, round to 32
    if (kv_end > 2112) kv_end = 2112;

    for (int kv = 0; kv < kv_end; kv += 32) {
        // Prefetch next K/V slab into cache (global_prefetch_b8) while we compute.
        if (kv + 32 < kv_end) {
            __builtin_prefetch(Kb + (size_t)(kv + 32 + lane) * 64, 0, 3);
            __builtin_prefetch(Vb + (size_t)(lane * 2) * 2112 + kv + 32, 0, 3);
        }

        // ---- S = Q K^T over the 32-col slab (two 16-col tiles) ----
        v8f s[2];
#pragma unroll
        for (int t = 0; t < 2; ++t) {
            const _Float16* kp = Kb + (size_t)(kv + t * 16 + colr) * 64 + halfu * 16;
            v16h bk0 = ld_frag(kp, kp + 8);        // d = halfu*16 + 0..15
            v16h bk1 = ld_frag(kp + 32, kp + 40);  // d = 32 + halfu*16 + 0..15
            v8f z = (v8f){0, 0, 0, 0, 0, 0, 0, 0};
            s[t] = wmma16(aq0, bk0, z);
            s[t] = wmma16(aq1, bk1, s[t]);
        }

        // ---- bias + online softmax (rows live in 16-lane halves) ----
        const int col0 = kv + colr;
        float p0a[8], p1a[8];
#pragma unroll
        for (int r = 0; r < 8; ++r) {
            const int qg = qrow0 + r + (halfu << 3);
            float s0 = s[0][r] + ((col0 < 64 || (col0 - 64) <= qg) ? 0.0f : NEG_BIAS);
            float s1 = s[1][r] + ((col0 < 48 || (col0 - 48) <= qg) ? 0.0f : NEG_BIAS);
            float rm = fmaxf(s0, s1);
            rm = fmaxf(rm, __shfl_xor(rm, 1));
            rm = fmaxf(rm, __shfl_xor(rm, 2));
            rm = fmaxf(rm, __shfl_xor(rm, 4));
            rm = fmaxf(rm, __shfl_xor(rm, 8));
            const float mnew = fmaxf(mstat[r], rm);
            const float scl = __expf(mstat[r] - mnew);
            const float p0 = __expf(s0 - mnew);
            const float p1 = __expf(s1 - mnew);
            float rs = p0 + p1;
            rs += __shfl_xor(rs, 1);
            rs += __shfl_xor(rs, 2);
            rs += __shfl_xor(rs, 4);
            rs += __shfl_xor(rs, 8);
            lstat[r] = lstat[r] * scl + rs;
            mstat[r] = mnew;
#pragma unroll
            for (int t = 0; t < 4; ++t) o[t][r] *= scl;
            p0a[r] = p0;
            p1a[r] = p1;
        }

        // ---- reshape P from C-layout to A-layout via per-wave LDS tile ----
#pragma unroll
        for (int r = 0; r < 8; ++r) {
            const int row = r + (halfu << 3);
            Pt[wave][row][colr] = (_Float16)p0a[r];
            Pt[wave][row][16 + colr] = (_Float16)p1a[r];
        }
        // Wave-private LDS region: DS ops are in-order within a wave.
        const _Float16* pb = &Pt[wave][colr][0];
        v16h apf = ld_frag(pb + halfu * 16, pb + halfu * 16 + 8);

        // ---- O += P V (V stored transposed [D, KT] -> contiguous B frags) ----
#pragma unroll
        for (int t = 0; t < 4; ++t) {
            const _Float16* vp = Vb + ((size_t)(t * 16 + colr)) * 2112 + kv + halfu * 16;
            v16h bv = ld_frag(vp, vp + 8);
            o[t] = wmma16(apf, bv, o[t]);
        }
    }

    // ---- finalize: normalize and write ctx [B, LQ, H*D] f16 ----
#pragma unroll
    for (int r = 0; r < 8; ++r) {
        const float inv = 1.0f / lstat[r];
        const int qg = qrow0 + r + (halfu << 3);
        _Float16* cp = Ctx + ((size_t)b * 2048 + qg) * 1024 + h * 64;
#pragma unroll
        for (int t = 0; t < 4; ++t) cp[t * 16 + colr] = (_Float16)(o[t][r] * inv);
    }
}

// ---------------------------------------------------------------------------
extern "C" void kernel_launch(void* const* d_in, const int* in_sizes, int n_in,
                              void* d_out, int out_size, void* d_ws, size_t ws_size,
                              hipStream_t stream) {
    const float* inputs_q     = (const float*)d_in[0];
    const float* inputs_kv    = (const float*)d_in[1];
    const float* key_prefix   = (const float*)d_in[2];
    const float* value_prefix = (const float*)d_in[3];
    // d_in[4] = causal mask (bool): never read, causality computed analytically.
    const float* Wq = (const float*)d_in[5];
    const float* Wk = (const float*)d_in[6];
    const float* Wv = (const float*)d_in[7];
    const float* Wo = (const float*)d_in[8];
    float* out = (float*)d_out;

    char* ws = (char*)d_ws;
    _Float16* Qh  = (_Float16*)(ws);             // [2,16,2048,64]  8.00 MB
    _Float16* Kh  = (_Float16*)(ws +  8388608);  // [2,16,2112,64]  8.25 MB
    _Float16* Vth = (_Float16*)(ws + 17039360);  // [2,16,64,2112]  8.25 MB
    _Float16* Ctx = (_Float16*)(ws + 25690112);  // [2,2048,1024]   8.00 MB

    const dim3 gemm_grid(32, 16);   // 4096/128 M-tiles x 1024/64 N-tiles
    const dim3 gemm_blk(256);

    // Prefix K/V -> f16 buffers (slots 0..63).
    prefix_copy<<<dim3(131072 / 256), dim3(256), 0, stream>>>(key_prefix, value_prefix,
                                                              Kh, Vth);
    // Q/K/V projections.
    gemm_qkvo<0><<<gemm_grid, gemm_blk, 0, stream>>>(inputs_q, nullptr, Wq, Qh, nullptr);
    gemm_qkvo<1><<<gemm_grid, gemm_blk, 0, stream>>>(inputs_kv, nullptr, Wk, Kh, nullptr);
    gemm_qkvo<2><<<gemm_grid, gemm_blk, 0, stream>>>(inputs_kv, nullptr, Wv, Vth, nullptr);
    // Flash attention: (LQ/64 q-tiles, H, B).
    flash_attn<<<dim3(32, 16, 2), dim3(128), 0, stream>>>(Qh, Kh, Vth, Ctx);
    // Output projection.
    gemm_qkvo<3><<<gemm_grid, gemm_blk, 0, stream>>>(nullptr, Ctx, Wo, nullptr, out);
}
